// ExtraEmbedding_49237505082085
// MI455X (gfx1250) — compile-verified, matrix operand order checked by reading, not run on hardware
//
#include <hip/hip_runtime.h>

// ---------------- types ----------------
typedef __attribute__((ext_vector_type(16))) __bf16 v16bf;
typedef __attribute__((ext_vector_type(8)))  float  v8f;
typedef __attribute__((ext_vector_type(4)))  unsigned int u32x4;
typedef int v4i_gcc __attribute__((__vector_size__(4 * sizeof(int))));

#define NB   4
#define LSEQ 2048
#define BL   8192            // NB*LSEQ
#define DMODEL 1024
#define DINNER 2048
#define NH   16
#define HD   128
#define DSTATE 64
#define CONVD 2176
#define DPROJ 4240
#define NC   32
#define CH   64

__device__ inline v8f v8f_zero() {
  v8f v = {0.f,0.f,0.f,0.f,0.f,0.f,0.f,0.f};
  return v;
}

// Load one 16x16x32 bf16 fragment row (32 K-contiguous elems at p).
// Per ISA layout: lanes 0-15 hold K {0..7, 16..23}; lanes 16-31 hold K {8..15, 24..31}.
__device__ inline v16bf frag_ld(const __bf16* p, int lane) {
  int kh = ((lane >> 4) & 1) * 8;
  union Fr { u32x4 u[2]; v16bf v; } t;
  t.u[0] = *(const u32x4*)(p + kh);
  t.u[1] = *(const u32x4*)(p + kh + 16);
  return t.v;
}

__device__ inline v8f wmma_bf16(v16bf a, v16bf b, v8f c) {
  return __builtin_amdgcn_wmma_f32_16x16x32_bf16(false, a, false, b, (short)0, c, false, false);
}

__device__ inline float silu_f(float x) { return x / (1.f + __expf(-x)); }

// ---------------- async global->LDS (CDNA5 GLOBAL_LOAD_ASYNC_TO_LDS_B128) -----
#if defined(__has_builtin)
#if __has_builtin(__builtin_amdgcn_global_load_async_to_lds_b128)
#define GEMM_ASYNC_LDS 1
#endif
#endif

#ifdef GEMM_ASYNC_LDS
__device__ inline void async_cp16(const void* g, void* l) {
  __builtin_amdgcn_global_load_async_to_lds_b128(
      (v4i_gcc*)const_cast<void*>(g), (v4i_gcc*)l, 0, 0);
}
__device__ inline void async_wait0() {
#if __has_builtin(__builtin_amdgcn_s_wait_asynccnt)
  __builtin_amdgcn_s_wait_asynccnt(0);
#else
  asm volatile("s_wait_asynccnt 0" ::: "memory");
#endif
}
#endif

// ---------------- 1) embed + LayerNorm -> bf16 ----------------
__global__ __launch_bounds__(256) void embed_ln_kernel(
    const float* __restrict__ x, const float* __restrict__ w,
    const float* __restrict__ g, const float* __restrict__ beta,
    __bf16* __restrict__ hb)
{
  int m = blockIdx.x;
  float xv = x[m];
  __shared__ float red[256];
  float s1 = 0.f, s2 = 0.f;
  for (int d = threadIdx.x; d < DMODEL; d += 256) {
    float wv = w[d]; s1 += wv; s2 += wv * wv;
  }
  red[threadIdx.x] = s1; __syncthreads();
  for (int s = 128; s > 0; s >>= 1) {
    if (threadIdx.x < s) red[threadIdx.x] += red[threadIdx.x + s];
    __syncthreads();
  }
  float Sw = red[0]; __syncthreads();
  red[threadIdx.x] = s2; __syncthreads();
  for (int s = 128; s > 0; s >>= 1) {
    if (threadIdx.x < s) red[threadIdx.x] += red[threadIdx.x + s];
    __syncthreads();
  }
  float Sw2 = red[0];
  float mu = xv * Sw * (1.f / DMODEL);
  float var = xv * xv * Sw2 * (1.f / DMODEL) - mu * mu;
  float rinv = rsqrtf(var + 1e-5f);
  for (int d = threadIdx.x; d < DMODEL; d += 256) {
    float hv = (xv * w[d] - mu) * rinv * g[d] + beta[d];
    hb[(size_t)m * DMODEL + d] = (__bf16)hv;
  }
}

// ---------------- f32 -> bf16 convert ----------------
__global__ __launch_bounds__(256) void f2bf_kernel(const float* __restrict__ in,
                                                   __bf16* __restrict__ out, int n)
{
  int i = blockIdx.x * 256 + threadIdx.x;
  if (i < n) out[i] = (__bf16)in[i];
}

// ---------------- 2) generic bf16 WMMA GEMM, double-buffered LDS ----------------
// A: MxK bf16 row-major, Bw: NxK bf16 row-major (weights), C: MxN f32
// Block tile 128x128, 8 waves, each wave 32x64 (2x4 wmma tiles), K-step 32.
// M and K must be multiples of 128/32 (true for all uses). GUARD_N handles the
// single partial column-tile of in_proj; full tiles take the unguarded path
// (async global->LDS when available, else register-staged with prefetch).
#define LDT 40   // 32 + 8 pad (80B rows, 16B-aligned)
template <bool GUARD_N>
__global__ __launch_bounds__(256) void gemm_bf16_kernel(
    const __bf16* __restrict__ A, const __bf16* __restrict__ Bw,
    float* __restrict__ C, int M, int N, int K, int n_tile0)
{
  __shared__ alignas(16) __bf16 As[2][128 * LDT];
  __shared__ alignas(16) __bf16 Bs[2][128 * LDT];
  const int tid = threadIdx.x;
  const int lane = tid & 31;
  const int wave = tid >> 5;
  const int wm = wave & 3;   // 0..3
  const int wn = wave >> 2;  // 0..1
  const int m0 = blockIdx.y * 128;
  const int n0 = (n_tile0 + blockIdx.x) * 128;

  // staging coordinates: thread stages rows r0 and r0+64 at element-col q8
  const int r0 = tid >> 2;         // 0..63
  const int q8 = (tid & 3) * 8;    // 0,8,16,24
  const __bf16* pa0 = A + (size_t)(m0 + r0) * K + q8;
  const __bf16* pa1 = pa0 + (size_t)64 * K;
  const __bf16* pb0 = Bw + (size_t)(n0 + r0) * K + q8;
  const __bf16* pb1 = pb0 + (size_t)64 * K;
  bool bv0 = true, bv1 = true;
  if (GUARD_N) { bv0 = (n0 + r0) < N; bv1 = (n0 + r0 + 64) < N; }

  const u32x4 zero4 = {0u, 0u, 0u, 0u};
  __bf16* as0 = &As[0][r0 * LDT + q8];
  __bf16* as1 = &As[0][(r0 + 64) * LDT + q8];
  __bf16* bs0 = &Bs[0][r0 * LDT + q8];
  __bf16* bs1 = &Bs[0][(r0 + 64) * LDT + q8];

  v8f acc[2][4];
  for (int i = 0; i < 2; i++)
    for (int j = 0; j < 4; j++) acc[i][j] = v8f_zero();

  const int nit = K >> 5;
  const int bufstride = 128 * LDT;   // elements per buffer

#ifdef GEMM_ASYNC_LDS
  if (!GUARD_N) {
    // ---- async direct global->LDS pipeline (ASYNCcnt-tracked, no VGPR staging)
    async_cp16(pa0, as0);
    async_cp16(pa1, as1);
    async_cp16(pb0, bs0);
    async_cp16(pb1, bs1);
    async_wait0();
    __syncthreads();
    for (int it = 0; it < nit; it++) {
      const int cur = it & 1;
      const int nxt = cur ^ 1;
      if (it + 1 < nit) {
        const int ko = (it + 1) * 32;
        async_cp16(pa0 + ko, as0 + nxt * bufstride);
        async_cp16(pa1 + ko, as1 + nxt * bufstride);
        async_cp16(pb0 + ko, bs0 + nxt * bufstride);
        async_cp16(pb1 + ko, bs1 + nxt * bufstride);
      }
      const __bf16* ab = &As[cur][0];
      const __bf16* bb = &Bs[cur][0];
      v16bf af[2], bfv[4];
      for (int mi = 0; mi < 2; mi++)
        af[mi] = frag_ld(ab + (wm * 32 + mi * 16 + (lane & 15)) * LDT, lane);
      for (int ni = 0; ni < 4; ni++)
        bfv[ni] = frag_ld(bb + (wn * 64 + ni * 16 + (lane & 15)) * LDT, lane);
      for (int mi = 0; mi < 2; mi++)
        for (int ni = 0; ni < 4; ni++)
          acc[mi][ni] = wmma_bf16(af[mi], bfv[ni], acc[mi][ni]);
      async_wait0();   // next buffer fully in LDS before publishing barrier
      __syncthreads();
    }
  } else
#endif
  {
    // ---- register-staged pipeline (edge tiles / fallback) ----
    *(u32x4*)as0 = *(const u32x4*)pa0;
    *(u32x4*)as1 = *(const u32x4*)pa1;
    *(u32x4*)bs0 = GUARD_N ? (bv0 ? *(const u32x4*)pb0 : zero4) : *(const u32x4*)pb0;
    *(u32x4*)bs1 = GUARD_N ? (bv1 ? *(const u32x4*)pb1 : zero4) : *(const u32x4*)pb1;
    __syncthreads();
    for (int it = 0; it < nit; it++) {
      const int cur = it & 1;
      const int nxt = cur ^ 1;
      const bool more = (it + 1) < nit;
      u32x4 ra0, ra1, rb0, rb1;
      if (more) {
        const int ko = (it + 1) * 32;
        ra0 = *(const u32x4*)(pa0 + ko);
        ra1 = *(const u32x4*)(pa1 + ko);
        rb0 = GUARD_N ? (bv0 ? *(const u32x4*)(pb0 + ko) : zero4) : *(const u32x4*)(pb0 + ko);
        rb1 = GUARD_N ? (bv1 ? *(const u32x4*)(pb1 + ko) : zero4) : *(const u32x4*)(pb1 + ko);
        if (it + 2 < nit) {   // stream hint two tiles ahead -> global_prefetch_b8
          const int ko2 = (it + 2) * 32;
          __builtin_prefetch(pa0 + ko2, 0, 0);
          __builtin_prefetch(pb0 + ko2, 0, 0);
        }
      }
      const __bf16* ab = &As[cur][0];
      const __bf16* bb = &Bs[cur][0];
      v16bf af[2], bfv[4];
      for (int mi = 0; mi < 2; mi++)
        af[mi] = frag_ld(ab + (wm * 32 + mi * 16 + (lane & 15)) * LDT, lane);
      for (int ni = 0; ni < 4; ni++)
        bfv[ni] = frag_ld(bb + (wn * 64 + ni * 16 + (lane & 15)) * LDT, lane);
      for (int mi = 0; mi < 2; mi++)
        for (int ni = 0; ni < 4; ni++)
          acc[mi][ni] = wmma_bf16(af[mi], bfv[ni], acc[mi][ni]);
      if (more) {
        *(u32x4*)(as0 + nxt * bufstride) = ra0;
        *(u32x4*)(as1 + nxt * bufstride) = ra1;
        *(u32x4*)(bs0 + nxt * bufstride) = rb0;
        *(u32x4*)(bs1 + nxt * bufstride) = rb1;
      }
      __syncthreads();
    }
  }

  for (int mi = 0; mi < 2; mi++)
    for (int ni = 0; ni < 4; ni++) {
      int nn = n0 + wn * 64 + ni * 16 + (lane & 15);
      if (GUARD_N && nn >= N) continue;
      int mb = m0 + wm * 32 + mi * 16 + ((lane >> 4) << 3);
      for (int r = 0; r < 8; r++)
        C[(size_t)(mb + r) * N + nn] = acc[mi][ni][r];
    }
}

// ---------------- 3a) dt = softplus(raw + bias) ----------------
__global__ __launch_bounds__(256) void dt_softplus_kernel(
    const float* __restrict__ zx, const float* __restrict__ dt_bias,
    float* __restrict__ gdt)
{
  int i = blockIdx.x * 256 + threadIdx.x;
  if (i >= BL * NH) return;
  int m = i >> 4, hh = i & 15;
  float v = zx[(size_t)m * DPROJ + (DINNER + CONVD) + hh] + dt_bias[hh];
  gdt[i] = (v > 20.f) ? v : log1pf(expf(v));
}

// ---------------- 3b) causal depthwise conv + silu ----------------
__global__ __launch_bounds__(256) void conv_silu_kernel(
    const float* __restrict__ zx, const float* __restrict__ cw,
    const float* __restrict__ cb, float* __restrict__ out)
{
  int i = blockIdx.x * 256 + threadIdx.x;
  if (i >= BL * CONVD) return;
  int cc = i % CONVD;
  int t = (i / CONVD) % LSEQ;
  int b = i / (CONVD * LSEQ);
  float a = cb[cc];
  for (int k = 0; k < 4; k++) {
    int tt = t - 3 + k;
    if (tt >= 0)
      a += zx[((size_t)(b * LSEQ + tt)) * DPROJ + DINNER + cc] * cw[cc * 4 + k];
  }
  out[i] = silu_f(a);
}

// ---------------- 4) SSD chunk kernel: Y_diag + per-chunk states ----------------
#define LDS_S 72  // 64 + 8 pad (144B rows, 16B-aligned)
__global__ __launch_bounds__(256) void ssd_chunk_kernel(
    const float* __restrict__ conv, const float* __restrict__ gdt,
    const float* __restrict__ A_log,
    float* __restrict__ gAcs, float* __restrict__ gSum,
    float* __restrict__ gStates, float* __restrict__ gY)
{
  const int c = blockIdx.x, h = blockIdx.y, b = blockIdx.z;
  const int tid = threadIdx.x, lane = tid & 31, wave = tid >> 5;
  __shared__ float sAcs[CH];
  __shared__ float sdt[CH];
  __shared__ alignas(16) __bf16 Ct[CH * LDS_S];   // [l][n]
  __shared__ alignas(16) __bf16 Bt[CH * LDS_S];   // [s][n]
  __shared__ alignas(16) __bf16 BTd[CH * LDS_S];  // [n][l] * decay_l
  __shared__ alignas(16) __bf16 Wt[CH * LDS_S];   // masked decayed G [l][s]
  __shared__ alignas(16) __bf16 XdT[HD * LDS_S];  // [p][s] (X * dt)

  const size_t rowbase = (size_t)(b * LSEQ + c * CH);
  if (tid < CH) sdt[tid] = gdt[(rowbase + tid) * NH + h];
  __syncthreads();
  if (tid == 0) {
    float Ah = -__expf(A_log[h]);
    float cs = 0.f;
    for (int l = 0; l < CH; l++) { cs += sdt[l] * Ah; sAcs[l] = cs; }
  }
  __syncthreads();
  const int cbase = (b * NH + h) * NC + c;
  if (tid < CH) gAcs[cbase * CH + tid] = sAcs[tid];
  if (tid == 0) gSum[cbase] = sAcs[CH - 1];
  float last = sAcs[CH - 1];

  for (int i = tid; i < CH * CH; i += 256) {
    int l = i >> 6, n = i & 63;
    const float* rowp = conv + (rowbase + l) * CONVD;
    float bv = rowp[DINNER + n];
    Ct[l * LDS_S + n] = (__bf16)rowp[DINNER + DSTATE + n];
    Bt[l * LDS_S + n] = (__bf16)bv;
    BTd[n * LDS_S + l] = (__bf16)(bv * __expf(last - sAcs[l]));
  }
  for (int i = tid; i < HD * CH; i += 256) {
    int p = i >> 6, s = i & 63;
    XdT[p * LDS_S + s] = (__bf16)(conv[(rowbase + s) * CONVD + h * HD + p] * sdt[s]);
  }
  __syncthreads();

  // GEMM1: G[l][s] = sum_n C[l,n]*B[s,n]; apply mask/decay -> Wt (bf16).
  // Branchless epilogue: exponent clamped to <=0 (valid entries already are),
  // masked entries multiplied by 0 (no inf*0 possible after the clamp).
  for (int t = wave * 2; t < wave * 2 + 2; t++) {
    int lt = t >> 2, st = t & 3;
    v8f acc = v8f_zero();
    for (int ks = 0; ks < CH; ks += 32) {
      v16bf a = frag_ld(Ct + (lt * 16 + (lane & 15)) * LDS_S + ks, lane);
      v16bf bb = frag_ld(Bt + (st * 16 + (lane & 15)) * LDS_S + ks, lane);
      acc = wmma_bf16(a, bb, acc);
    }
    int l0 = lt * 16 + ((lane >> 4) << 3);
    int s = st * 16 + (lane & 15);
    float acs_s = sAcs[s];
    for (int r = 0; r < 8; r++) {
      int ll = l0 + r;
      float e = fminf(sAcs[ll] - acs_s, 0.f);
      float msk = (s <= ll) ? 1.f : 0.f;
      Wt[ll * LDS_S + s] = (__bf16)(msk * acc[r] * __expf(e));
    }
  }
  __syncthreads();

  // GEMM2: Y_diag[l][p] = sum_s W[l,s]*Xd[s,p]
  for (int i = 0; i < 4; i++) {
    int t = wave * 4 + i;
    int lt = t >> 3, pt = t & 7;
    v8f acc = v8f_zero();
    for (int ks = 0; ks < CH; ks += 32) {
      v16bf a = frag_ld(Wt + (lt * 16 + (lane & 15)) * LDS_S + ks, lane);
      v16bf bb = frag_ld(XdT + (pt * 16 + (lane & 15)) * LDS_S + ks, lane);
      acc = wmma_bf16(a, bb, acc);
    }
    int l0 = lt * 16 + ((lane >> 4) << 3);
    int p = pt * 16 + (lane & 15);
    for (int r = 0; r < 8; r++)
      gY[(rowbase + l0 + r) * DINNER + h * HD + p] = acc[r];
  }

  // GEMM3: states[n][p] = sum_l B[l,n]*decay_l*Xd[l,p]
  for (int i = 0; i < 4; i++) {
    int t = wave * 4 + i;
    int nt = t >> 3, pt = t & 7;
    v8f acc = v8f_zero();
    for (int ks = 0; ks < CH; ks += 32) {
      v16bf a = frag_ld(BTd + (nt * 16 + (lane & 15)) * LDS_S + ks, lane);
      v16bf bb = frag_ld(XdT + (pt * 16 + (lane & 15)) * LDS_S + ks, lane);
      acc = wmma_bf16(a, bb, acc);
    }
    int n0 = nt * 16 + ((lane >> 4) << 3);
    int p = pt * 16 + (lane & 15);
    for (int r = 0; r < 8; r++)
      gStates[(size_t)cbase * (DSTATE * HD) + (n0 + r) * HD + p] = acc[r];
  }
}

// ---------------- 5) inter-chunk scan (in-place: states[c] -> prev_states[c]) ----
__global__ __launch_bounds__(256) void ssd_scan_kernel(
    const float* __restrict__ gSum, float* __restrict__ gStates)
{
  int h = blockIdx.x, b = blockIdx.y;
  int bh = b * NH + h;
  float run[32];
  for (int i = 0; i < 32; i++) run[i] = 0.f;
  for (int c = 0; c < NC; c++) {
    float dec = __expf(gSum[bh * NC + c]);
    float* st = gStates + (size_t)(bh * NC + c) * (DSTATE * HD);
    for (int i = 0; i < 32; i++) {
      int idx = threadIdx.x + i * 256;
      float tmp = st[idx];
      st[idx] = run[i];
      run[i] = run[i] * dec + tmp;
    }
  }
}

// ---------------- 6) Y_off = exp(Acs)*C.prev_states; Y += Y_off + X*D ----------
__global__ __launch_bounds__(256) void ssd_yoff_kernel(
    const float* __restrict__ conv, const float* __restrict__ gAcs,
    const float* __restrict__ gStates, const float* __restrict__ Dparam,
    float* __restrict__ gY)
{
  const int c = blockIdx.x, h = blockIdx.y, b = blockIdx.z;
  const int tid = threadIdx.x, lane = tid & 31, wave = tid >> 5;
  __shared__ float sAcs[CH];
  __shared__ alignas(16) __bf16 Ct[CH * LDS_S];  // [l][n]
  __shared__ alignas(16) __bf16 PT[HD * LDS_S];  // [p][n]
  const size_t rowbase = (size_t)(b * LSEQ + c * CH);
  const int cbase = (b * NH + h) * NC + c;
  if (tid < CH) sAcs[tid] = gAcs[cbase * CH + tid];
  for (int i = tid; i < CH * CH; i += 256) {
    int l = i >> 6, n = i & 63;
    Ct[l * LDS_S + n] = (__bf16)conv[(rowbase + l) * CONVD + DINNER + DSTATE + n];
  }
  for (int i = tid; i < HD * CH; i += 256) {
    int p = i >> 6, n = i & 63;
    PT[p * LDS_S + n] = (__bf16)gStates[(size_t)cbase * (DSTATE * HD) + n * HD + p];
  }
  __syncthreads();
  float Dh = Dparam[h];
  for (int i = 0; i < 4; i++) {
    int t = wave * 4 + i;
    int lt = t >> 3, pt = t & 7;
    v8f acc = v8f_zero();
    for (int ks = 0; ks < CH; ks += 32) {
      v16bf a = frag_ld(Ct + (lt * 16 + (lane & 15)) * LDS_S + ks, lane);
      v16bf bb = frag_ld(PT + (pt * 16 + (lane & 15)) * LDS_S + ks, lane);
      acc = wmma_bf16(a, bb, acc);
    }
    int l0 = lt * 16 + ((lane >> 4) << 3);
    int p = pt * 16 + (lane & 15);
    for (int r = 0; r < 8; r++) {
      int l = l0 + r;
      size_t row = rowbase + l;
      float xv = conv[row * CONVD + h * HD + p];
      size_t yi = row * DINNER + h * HD + p;
      gY[yi] = gY[yi] + acc[r] * __expf(sAcs[l]) + xv * Dh;
    }
  }
}

// ---------------- 7) gate (silu(z)) + RMSNorm -> bf16 ----------------
__global__ __launch_bounds__(256) void gate_rms_kernel(
    const float* __restrict__ gY, const float* __restrict__ zx,
    const float* __restrict__ rms_w, __bf16* __restrict__ yb)
{
  int m = blockIdx.x;
  float yv[8];
  float ss = 0.f;
  for (int i = 0; i < 8; i++) {
    int d = threadIdx.x + i * 256;
    float z = zx[(size_t)m * DPROJ + d];
    float y = gY[(size_t)m * DINNER + d] * silu_f(z);
    yv[i] = y; ss += y * y;
  }
  __shared__ float red[256];
  red[threadIdx.x] = ss; __syncthreads();
  for (int s = 128; s > 0; s >>= 1) {
    if (threadIdx.x < s) red[threadIdx.x] += red[threadIdx.x + s];
    __syncthreads();
  }
  float rinv = rsqrtf(red[0] * (1.f / DINNER) + 1e-5f);
  for (int i = 0; i < 8; i++) {
    int d = threadIdx.x + i * 256;
    yb[(size_t)m * DINNER + d] = (__bf16)(yv[i] * rinv * rms_w[d]);
  }
}

// ---------------- launch ----------------
extern "C" void kernel_launch(void* const* d_in, const int* in_sizes, int n_in,
                              void* d_out, int out_size, void* d_ws, size_t ws_size,
                              hipStream_t stream) {
  (void)in_sizes; (void)n_in; (void)out_size; (void)ws_size;
  const float* x       = (const float*)d_in[0];
  const float* w_embed = (const float*)d_in[1];
  const float* ln_g    = (const float*)d_in[2];
  const float* ln_b    = (const float*)d_in[3];
  const float* W_in    = (const float*)d_in[4];
  const float* conv_w  = (const float*)d_in[5];
  const float* conv_b  = (const float*)d_in[6];
  const float* dt_bias = (const float*)d_in[7];
  const float* A_log   = (const float*)d_in[8];
  const float* Dp      = (const float*)d_in[9];
  const float* rms_w   = (const float*)d_in[10];
  const float* W_out   = (const float*)d_in[11];
  float* out = (float*)d_out;

  char* ws = (char*)d_ws;
  size_t off = 0;
  auto alloc = [&](size_t bytes) {
    size_t o = off; off += (bytes + 255) & ~(size_t)255; return o;
  };
  __bf16* hb      = (__bf16*)(ws + alloc((size_t)BL * DMODEL * 2));
  __bf16* Winb    = (__bf16*)(ws + alloc((size_t)DPROJ * DMODEL * 2));
  __bf16* Woutb   = (__bf16*)(ws + alloc((size_t)DMODEL * DINNER * 2));
  float*  zx      = (float*)(ws + alloc((size_t)BL * DPROJ * 4));
  float*  conv    = (float*)(ws + alloc((size_t)BL * CONVD * 4));
  float*  gdt     = (float*)(ws + alloc((size_t)BL * NH * 4));
  float*  gAcs    = (float*)(ws + alloc((size_t)NB * NH * NC * CH * 4));
  float*  gSum    = (float*)(ws + alloc((size_t)NB * NH * NC * 4));
  float*  gStates = (float*)(ws + alloc((size_t)NB * NH * NC * DSTATE * HD * 4));
  float*  gY      = (float*)(ws + alloc((size_t)BL * DINNER * 4));
  __bf16* yb      = (__bf16*)(ws + alloc((size_t)BL * DINNER * 2));

  embed_ln_kernel<<<BL, 256, 0, stream>>>(x, w_embed, ln_g, ln_b, hb);
  f2bf_kernel<<<(DPROJ * DMODEL + 255) / 256, 256, 0, stream>>>(W_in, Winb, DPROJ * DMODEL);
  f2bf_kernel<<<(DMODEL * DINNER + 255) / 256, 256, 0, stream>>>(W_out, Woutb, DMODEL * DINNER);

  // in_proj: (8192 x 1024) x (4240 x 1024)^T -> (8192 x 4240)
  // 33 full column tiles unguarded + 1 guarded edge tile (4240 = 33*128 + 16)
  gemm_bf16_kernel<false><<<dim3(DPROJ / 128, BL / 128), 256, 0, stream>>>(
      hb, Winb, zx, BL, DPROJ, DMODEL, 0);
  gemm_bf16_kernel<true><<<dim3(1, BL / 128), 256, 0, stream>>>(
      hb, Winb, zx, BL, DPROJ, DMODEL, DPROJ / 128);

  dt_softplus_kernel<<<(BL * NH + 255) / 256, 256, 0, stream>>>(zx, dt_bias, gdt);
  conv_silu_kernel<<<(BL * CONVD + 255) / 256, 256, 0, stream>>>(zx, conv_w, conv_b, conv);

  ssd_chunk_kernel<<<dim3(NC, NH, NB), 256, 0, stream>>>(
      conv, gdt, A_log, gAcs, gSum, gStates, gY);
  ssd_scan_kernel<<<dim3(NH, NB), 256, 0, stream>>>(gSum, gStates);
  ssd_yoff_kernel<<<dim3(NC, NH, NB), 256, 0, stream>>>(conv, gAcs, gStates, Dp, gY);

  gate_rms_kernel<<<BL, 256, 0, stream>>>(gY, zx, rms_w, yb);

  // out_proj: (8192 x 2048) x (1024 x 2048)^T -> (8192 x 1024), all tiles full
  gemm_bf16_kernel<false><<<dim3(DMODEL / 128, BL / 128), 256, 0, stream>>>(
      yb, Woutb, out, BL, DMODEL, DINNER, 0);
}
